// GCN_Module_4045859193694
// MI455X (gfx1250) — matrix operand order repaired and checked by reading, not compiled, not run on hardware
//
#include <hip/hip_runtime.h>

// ---------------- problem constants ----------------
#define NB   128          // batch
#define NN   128          // boxes per batch
#define NFG  1024
#define NFR  256
#define NG   16
#define BN   (NB*NN)      // 16384
#define OUT0 (BN*NFG)     // 16777216 floats: "out" part of d_out
#define LN_EPS 1e-5f
#define POS_THR (0.2f*157.0f)   // 31.4

// ---------------- WMMA plumbing (CDNA5 wave32) ----------------
typedef __attribute__((ext_vector_type(16))) __bf16        bf16x16;
typedef __attribute__((ext_vector_type(8)))  float         f32x8;
typedef __attribute__((ext_vector_type(8)))  unsigned short u16x8;

union FragU { bf16x16 v; u16x8 u[2]; };

__device__ __forceinline__ unsigned short f2bf(float x) {
  unsigned int u = __float_as_uint(x);
  u += 0x7fffu + ((u >> 16) & 1u);          // round-to-nearest-even
  return (unsigned short)(u >> 16);
}

// A fragment: 16x32 (MxK), row-major bf16 source with leading dim ld.
// lane l: row m0+(l&15); K = {k0+8h+j | j<8} U {k0+16+8h+j}
__device__ __forceinline__ bf16x16 frag_a(const unsigned short* base, int m0, int k0, int ld, int lane) {
  const unsigned short* p = base + (m0 + (lane & 15)) * ld + k0 + 8 * (lane >> 4);
  FragU f;
  f.u[0] = *(const u16x8*)p;
  f.u[1] = *(const u16x8*)(p + 16);
  return f.v;
}

// B fragment: 32x16 (KxN) where B[k][n] = W[n][k], W row-major with leading dim ld.
// lane l: col n0+(l&15); K = {k0+8h+j | j<8} U {k0+16+8h+j}
__device__ __forceinline__ bf16x16 frag_b(const unsigned short* base, int n0, int k0, int ld, int lane) {
  const unsigned short* p = base + (n0 + (lane & 15)) * ld + k0 + 8 * (lane >> 4);
  FragU f;
  f.u[0] = *(const u16x8*)p;
  f.u[1] = *(const u16x8*)(p + 16);
  return f.v;
}

#define WMMA_BF16(a, b, c) \
  __builtin_amdgcn_wmma_f32_16x16x32_bf16(false, (a), false, (b), (short)0, (c), false, false)

// ---------------- kernel 0: x -> bf16, box centers ----------------
__global__ void k_prep(const float* __restrict__ x, const float* __restrict__ boxes,
                       unsigned short* __restrict__ xbf, float* __restrict__ centers) {
  int tid = blockIdx.x * 256 + threadIdx.x;
  int stride = gridDim.x * 256;
  for (int e = tid; e < OUT0; e += stride) xbf[e] = f2bf(x[e]);
  if (tid < BN) {
    centers[2 * tid + 0] = 0.5f * (boxes[4 * tid + 0] + boxes[4 * tid + 2]);
    centers[2 * tid + 1] = 0.5f * (boxes[4 * tid + 1] + boxes[4 * tid + 3]);
  }
}

// ---------------- kernel W: per-group weights -> bf16 ----------------
__global__ void k_wconv(const float* __restrict__ thw, const float* __restrict__ phw,
                        const float* __restrict__ gcw,
                        unsigned short* __restrict__ thwb, unsigned short* __restrict__ phwb,
                        unsigned short* __restrict__ gcwb) {
  int tid = blockIdx.x * 256 + threadIdx.x;
  int stride = gridDim.x * 256;
  for (int e = tid; e < NFR * NFG; e += stride) { thwb[e] = f2bf(thw[e]); phwb[e] = f2bf(phw[e]); }
  for (int e = tid; e < NFG * NFG; e += stride) gcwb[e] = f2bf(gcw[e]);
}

// ---------------- kernel 1: th/ph = x @ W^T + b  (M=16384,K=1024,N=256) ----------------
// grid (256, 1, 2): z=0 -> theta, z=1 -> phi.  8 waves/block, wave -> (m-tile, n-half).
// K-loop ping-pong double buffered: two K=32 steps per iteration.
__global__ void __launch_bounds__(256, 1)
k_gemm1(const unsigned short* __restrict__ xbf,
        const unsigned short* __restrict__ thwb, const unsigned short* __restrict__ phwb,
        const float* __restrict__ thb, const float* __restrict__ phb,
        unsigned short* __restrict__ th, unsigned short* __restrict__ ph) {
  const unsigned short* w = blockIdx.z ? phwb : thwb;
  const float* bias = blockIdx.z ? phb : thb;
  unsigned short* dst = blockIdx.z ? ph : th;
  int lane = threadIdx.x & 31, wave = threadIdx.x >> 5;
  int m0 = (blockIdx.x * 4 + (wave >> 1)) * 16;
  int nbase = (wave & 1) * 8;
  f32x8 acc[8] = {};
  bf16x16 aA, aB, bA[8], bB[8];
  aA = frag_a(xbf, m0, 0, NFG, lane);
#pragma unroll
  for (int nt = 0; nt < 8; ++nt) bA[nt] = frag_b(w, (nbase + nt) * 16, 0, NFG, lane);
  for (int k0 = 0; k0 < NFG; k0 += 64) {
    // prefetch set B (k0+32)
    aB = frag_a(xbf, m0, k0 + 32, NFG, lane);
#pragma unroll
    for (int nt = 0; nt < 8; ++nt) bB[nt] = frag_b(w, (nbase + nt) * 16, k0 + 32, NFG, lane);
    // compute set A (k0)
#pragma unroll
    for (int nt = 0; nt < 8; ++nt) acc[nt] = WMMA_BF16(aA, bA[nt], acc[nt]);
    // prefetch set A for next iteration (clamped; last prefetch discarded)
    int kn = (k0 + 64 < NFG) ? k0 + 64 : 0;
    aA = frag_a(xbf, m0, kn, NFG, lane);
#pragma unroll
    for (int nt = 0; nt < 8; ++nt) bA[nt] = frag_b(w, (nbase + nt) * 16, kn, NFG, lane);
    // compute set B (k0+32)
#pragma unroll
    for (int nt = 0; nt < 8; ++nt) acc[nt] = WMMA_BF16(aB, bB[nt], acc[nt]);
  }
  int col = lane & 15, rb = 8 * (lane >> 4);
#pragma unroll
  for (int nt = 0; nt < 8; ++nt) {
    int n0 = (nbase + nt) * 16;
    float bv = bias[n0 + col];
#pragma unroll
    for (int v = 0; v < 8; ++v)
      dst[(m0 + rb + v) * NFR + n0 + col] = f2bf(acc[nt][v] + bv);
  }
}

// ---------------- kernel 2: sim = th·ph^T /16, mask, softmax -> rg ----------------
// one block per batch; 8 waves each own a 16-row stripe of the 128x128 sim.
__global__ void __launch_bounds__(256, 1)
k_sim(const unsigned short* __restrict__ th, const unsigned short* __restrict__ ph,
      const float* __restrict__ centers,
      float* __restrict__ rg_out, unsigned short* __restrict__ rgbf) {
  __shared__ float sim_s[128 * 132];
  int b = blockIdx.x;
  int lane = threadIdx.x & 31, wave = threadIdx.x >> 5;
  const unsigned short* thp = th + b * NN * NFR;
  const unsigned short* php = ph + b * NN * NFR;
  int m0 = wave * 16;
  f32x8 acc[8] = {};
  bf16x16 aA, aB, bA[8], bB[8];
  aA = frag_a(thp, m0, 0, NFR, lane);
#pragma unroll
  for (int nt = 0; nt < 8; ++nt) bA[nt] = frag_b(php, nt * 16, 0, NFR, lane);
  for (int k0 = 0; k0 < NFR; k0 += 64) {
    aB = frag_a(thp, m0, k0 + 32, NFR, lane);
#pragma unroll
    for (int nt = 0; nt < 8; ++nt) bB[nt] = frag_b(php, nt * 16, k0 + 32, NFR, lane);
#pragma unroll
    for (int nt = 0; nt < 8; ++nt) acc[nt] = WMMA_BF16(aA, bA[nt], acc[nt]);
    int kn = (k0 + 64 < NFR) ? k0 + 64 : 0;
    aA = frag_a(thp, m0, kn, NFR, lane);
#pragma unroll
    for (int nt = 0; nt < 8; ++nt) bA[nt] = frag_b(php, nt * 16, kn, NFR, lane);
#pragma unroll
    for (int nt = 0; nt < 8; ++nt) acc[nt] = WMMA_BF16(aB, bB[nt], acc[nt]);
  }
  int col = lane & 15, rb = 8 * (lane >> 4);
#pragma unroll
  for (int nt = 0; nt < 8; ++nt)
#pragma unroll
    for (int v = 0; v < 8; ++v)
      sim_s[(m0 + rb + v) * 132 + nt * 16 + col] = acc[nt][v] * 0.0625f;  // 1/sqrt(256)
  __syncthreads();
  if (threadIdx.x < 128) {
    int n = threadIdx.x;
    float cx = centers[(b * NN + n) * 2 + 0];
    float cy = centers[(b * NN + n) * 2 + 1];
    const float ninf = -__builtin_huge_valf();
    float mx = ninf;
    for (int m = 0; m < NN; ++m) {
      float dx = cx - centers[(b * NN + m) * 2 + 0];
      float dy = cy - centers[(b * NN + m) * 2 + 1];
      float s = (dx * dx + dy * dy > POS_THR * POS_THR) ? ninf : sim_s[n * 132 + m];
      sim_s[n * 132 + m] = s;
      mx = fmaxf(mx, s);
    }
    float sum = 0.f;
    for (int m = 0; m < NN; ++m) {
      float e = __expf(sim_s[n * 132 + m] - mx);
      sim_s[n * 132 + m] = e;
      sum += e;
    }
    float inv = 1.0f / sum;
    for (int m = 0; m < NN; ++m) {
      float r = sim_s[n * 132 + m] * inv;
      rg_out[(b * NN + n) * NN + m] = r;          // relation_graph (f32, d_out tail)
      rgbf[(b * NN + n) * NN + m] = f2bf(r);      // bf16 for next WMMA
    }
  }
}

// ---------------- kernel 3: agg = rg @ x   (per batch: 128x1024, K=128) ----------------
// grid (8 f-chunks, 128 batches); x chunk transposed into LDS so B is column-major.
__global__ void __launch_bounds__(256, 1)
k_agg(const unsigned short* __restrict__ rgbf, const unsigned short* __restrict__ xbf,
      unsigned short* __restrict__ agg) {
  __shared__ unsigned short xT[128 * 136];   // [f][m], padded stride
  int b = blockIdx.y;
  int f0 = blockIdx.x * 128;
  for (int e = threadIdx.x; e < 128 * 128; e += 256) {
    int m = e >> 7, f = e & 127;
    xT[f * 136 + m] = xbf[(b * NN + m) * NFG + f0 + f];
  }
  __syncthreads();
  int lane = threadIdx.x & 31, wave = threadIdx.x >> 5;
  int n0 = wave * 16;
  const unsigned short* rgb = rgbf + b * NN * NN;
  f32x8 acc[8] = {};
  bf16x16 aA, aB, bA[8], bB[8];
  aA = frag_a(rgb, n0, 0, NN, lane);
#pragma unroll
  for (int ft = 0; ft < 8; ++ft) bA[ft] = frag_b(xT, ft * 16, 0, 136, lane);
  for (int k0 = 0; k0 < NN; k0 += 64) {
    aB = frag_a(rgb, n0, k0 + 32, NN, lane);
#pragma unroll
    for (int ft = 0; ft < 8; ++ft) bB[ft] = frag_b(xT, ft * 16, k0 + 32, 136, lane);
#pragma unroll
    for (int ft = 0; ft < 8; ++ft) acc[ft] = WMMA_BF16(aA, bA[ft], acc[ft]);
    int kn = (k0 + 64 < NN) ? k0 + 64 : 0;
    aA = frag_a(rgb, n0, kn, NN, lane);
#pragma unroll
    for (int ft = 0; ft < 8; ++ft) bA[ft] = frag_b(xT, ft * 16, kn, 136, lane);
#pragma unroll
    for (int ft = 0; ft < 8; ++ft) acc[ft] = WMMA_BF16(aB, bB[ft], acc[ft]);
  }
  int col = lane & 15, rb = 8 * (lane >> 4);
#pragma unroll
  for (int ft = 0; ft < 8; ++ft)
#pragma unroll
    for (int v = 0; v < 8; ++v)
      agg[(b * NN + n0 + rb + v) * NFG + f0 + ft * 16 + col] = f2bf(acc[ft][v]);
}

// ---------------- kernel 4: h = agg @ gcn_w^T, LayerNorm, ReLU, out += ----------------
// one block = one 16-row stripe (full 1024 cols so LN sees whole rows).
__global__ void __launch_bounds__(256, 1)
k_gemm2ln(const unsigned short* __restrict__ agg, const unsigned short* __restrict__ gcwb,
          const float* __restrict__ lng, const float* __restrict__ lnb,
          float* __restrict__ out, int g) {
  __shared__ float h_s[16 * 1024];
  __shared__ float red[16 * 16 * 2];
  __shared__ float stat[16 * 2];
  int m0 = blockIdx.x * 16;
  int lane = threadIdx.x & 31, wave = threadIdx.x >> 5;
  f32x8 acc[8] = {};
  bf16x16 aA, aB, bA[8], bB[8];
  aA = frag_a(agg, m0, 0, NFG, lane);
#pragma unroll
  for (int nt = 0; nt < 8; ++nt) bA[nt] = frag_b(gcwb, (wave * 8 + nt) * 16, 0, NFG, lane);
  for (int k0 = 0; k0 < NFG; k0 += 64) {
    aB = frag_a(agg, m0, k0 + 32, NFG, lane);
#pragma unroll
    for (int nt = 0; nt < 8; ++nt) bB[nt] = frag_b(gcwb, (wave * 8 + nt) * 16, k0 + 32, NFG, lane);
#pragma unroll
    for (int nt = 0; nt < 8; ++nt) acc[nt] = WMMA_BF16(aA, bA[nt], acc[nt]);
    int kn = (k0 + 64 < NFG) ? k0 + 64 : 0;
    aA = frag_a(agg, m0, kn, NFG, lane);
#pragma unroll
    for (int nt = 0; nt < 8; ++nt) bA[nt] = frag_b(gcwb, (wave * 8 + nt) * 16, kn, NFG, lane);
#pragma unroll
    for (int nt = 0; nt < 8; ++nt) acc[nt] = WMMA_BF16(aB, bB[nt], acc[nt]);
  }
  int col = lane & 15, rb = 8 * (lane >> 4);
#pragma unroll
  for (int nt = 0; nt < 8; ++nt)
#pragma unroll
    for (int v = 0; v < 8; ++v)
      h_s[(rb + v) * 1024 + (wave * 8 + nt) * 16 + col] = acc[nt][v];
  __syncthreads();
  // LayerNorm stats: 16 threads per row
  int row = threadIdx.x >> 4, sub = threadIdx.x & 15;
  float s = 0.f, s2 = 0.f;
  for (int k = sub; k < 1024; k += 16) { float hv = h_s[row * 1024 + k]; s += hv; s2 += hv * hv; }
  red[(row * 16 + sub) * 2 + 0] = s;
  red[(row * 16 + sub) * 2 + 1] = s2;
  __syncthreads();
  if (sub == 0) {
    float ts = 0.f, ts2 = 0.f;
    for (int k = 0; k < 16; ++k) { ts += red[(row * 16 + k) * 2]; ts2 += red[(row * 16 + k) * 2 + 1]; }
    float mu = ts * (1.0f / 1024.0f);
    float var = ts2 * (1.0f / 1024.0f) - mu * mu;
    stat[row * 2 + 0] = mu;
    stat[row * 2 + 1] = rsqrtf(var + LN_EPS);
  }
  __syncthreads();
  for (int i = 0; i < 64; ++i) {
    int e = threadIdx.x + 256 * i;        // 0..16383
    int r = e >> 10, c = e & 1023;
    float hv = (h_s[r * 1024 + c] - stat[r * 2]) * stat[r * 2 + 1] * lng[c] + lnb[c];
    hv = fmaxf(hv, 0.f);
    int oi = (m0 + r) * 1024 + c;
    if (g == 0) out[oi] = hv; else out[oi] += hv;
  }
}

// ---------------- host launcher ----------------
extern "C" void kernel_launch(void* const* d_in, const int* in_sizes, int n_in,
                              void* d_out, int out_size, void* d_ws, size_t ws_size,
                              hipStream_t stream) {
  const float* x      = (const float*)d_in[0];
  const float* boxes  = (const float*)d_in[1];
  const float* thw    = (const float*)d_in[2];
  const float* thb    = (const float*)d_in[3];
  const float* phw    = (const float*)d_in[4];
  const float* phb    = (const float*)d_in[5];
  const float* gcw    = (const float*)d_in[6];
  const float* lng    = (const float*)d_in[7];
  const float* lnb    = (const float*)d_in[8];
  float* out   = (float*)d_out;
  float* rgout = out + OUT0;               // relation_graph output (last group wins)

  char* ws = (char*)d_ws;
  unsigned short* xbf    = (unsigned short*)(ws);                         // 32 MB
  float*          cent   = (float*)(ws + 33554432);                      // 128 KB
  unsigned short* thwb   = (unsigned short*)(ws + 33685504);             // 512 KB
  unsigned short* phwb   = (unsigned short*)(ws + 34209792);             // 512 KB
  unsigned short* gcwb   = (unsigned short*)(ws + 34734080);             // 2 MB
  unsigned short* th     = (unsigned short*)(ws + 36831232);             // 8 MB
  unsigned short* ph     = (unsigned short*)(ws + 45219840);             // 8 MB
  unsigned short* rgbf   = (unsigned short*)(ws + 53608448);             // 4 MB
  unsigned short* agg    = (unsigned short*)(ws + 57802752);             // 32 MB

  k_prep<<<4096, 256, 0, stream>>>(x, boxes, xbf, cent);
  for (int g = 0; g < NG; ++g) {
    k_wconv<<<1024, 256, 0, stream>>>(thw + (size_t)g * NFR * NFG, phw + (size_t)g * NFR * NFG,
                                      gcw + (size_t)g * NFG * NFG, thwb, phwb, gcwb);
    k_gemm1<<<dim3(256, 1, 2), 256, 0, stream>>>(xbf, thwb, phwb,
                                                 thb + g * NFR, phb + g * NFR, th, ph);
    k_sim<<<NB, 256, 0, stream>>>(th, ph, cent, rgout, rgbf);
    k_agg<<<dim3(8, NB), 256, 0, stream>>>(rgbf, xbf, agg);
    k_gemm2ln<<<BN / 16, 256, 0, stream>>>(agg, gcwb, lng + g * NFG, lnb + g * NFG, out, g);
  }
  (void)in_sizes; (void)n_in; (void)out_size; (void)ws_size;
}
